// URHMModel_85718957294322
// MI455X (gfx1250) — compile-verified, matrix operand order checked by reading, not run on hardware
//
#include <hip/hip_runtime.h>
#include <hip/hip_fp16.h>
#include <math.h>

// MI455X / gfx1250, wave32, WMMA 16x16x32 f16->f32.
//
// Fused conv(1x1) + 120-step tanh RNN (H=256, B=8192) + softmax.
// One WG = 128 batch rows, 512 threads (16 waves). Each wave owns 2 N-tiles
// and keeps all its w_hh / w_ih B-fragments resident in VGPRs for the whole
// time loop; hidden state ping-pongs through LDS as f16; accumulation in f32.

typedef __attribute__((ext_vector_type(16))) _Float16 v16h;
typedef __attribute__((ext_vector_type(8)))  float    v8f;

#define T_SEQ    120
#define HDIM     256
#define BT       128      // batch rows per workgroup
#define NTHREADS 512      // 16 waves
#define KIN      21       // input-projection K (padded to 32)

// Dynamic LDS layout (bytes):
//   [0)          hbuf : 2 * BT * HDIM  halves = 131072
//   [131072)     ebuf : 2 * BT * 32    halves =  16384
//   [147456)     bias : HDIM floats            =   1024
#define SMEM_BYTES 148480

// ---------------------------------------------------------------------------
// A-fragment (16x32 f16, M x K) from an f16 LDS matrix, row stride in halves.
// ISA layout: lanes 0-15 -> row M=lane, halves = K{k0..k0+7, k0+16..k0+23};
//             lanes 16-31 -> row M=lane-16, halves = K{k0+8..15, k0+24..31}.
__device__ __forceinline__ v16h lds_fragA(const _Float16* base, int row0,
                                          int strideH, int k0, int lane) {
  int r  = row0 + (lane & 15);
  int kb = k0 + ((lane & 16) ? 8 : 0);
  const _Float16* p = base + r * strideH + kb;
  union { v16h v; uint4 u[2]; } f;
  f.u[0] = *reinterpret_cast<const uint4*>(p);        // ds_load_b128
  f.u[1] = *reinterpret_cast<const uint4*>(p + 16);   // ds_load_b128
  return f.v;
}

// B-fragment (32x16 f16, K x N) with B[k][n] = W[n][k], W f32 row-major in
// global memory (prologue only; converts f32 -> f16 per element).
__device__ __forceinline__ v16h glob_fragB_whh(const float* __restrict__ W,
                                               int n0, int k0, int lane) {
  int n  = n0 + (lane & 15);
  int kb = k0 + ((lane & 16) ? 8 : 0);
  const float* p = W + n * HDIM + kb;
  v16h v;
#pragma unroll
  for (int j = 0; j < 8; ++j) v[j]     = (_Float16)p[j];
#pragma unroll
  for (int j = 0; j < 8; ++j) v[8 + j] = (_Float16)p[16 + j];
  return v;
}

// Same, for w_ih (HDIM x 21, row-major) zero-padded to K=32.
__device__ __forceinline__ v16h glob_fragB_wih(const float* __restrict__ W,
                                               int n0, int lane) {
  int n  = n0 + (lane & 15);
  int kb = (lane & 16) ? 8 : 0;
  v16h v;
#pragma unroll
  for (int j = 0; j < 8; ++j) {
    int k = kb + j;
    v[j] = (k < KIN) ? (_Float16)W[n * KIN + k] : (_Float16)0.0f;
  }
#pragma unroll
  for (int j = 0; j < 8; ++j) {
    int k = kb + 16 + j;
    v[8 + j] = (k < KIN) ? (_Float16)W[n * KIN + k] : (_Float16)0.0f;
  }
  return v;
}

// tanh = 1 - 2*rcp(e^{2x}+1) with v_exp_f32 + v_rcp_f32 (both TRANS ops,
// co-execute with VALU). Saturates: e=inf -> rcp(inf)=0 -> +1; e=0 -> -1.
__device__ __forceinline__ float fast_tanh(float x) {
  float e = __expf(2.0f * x);
  return 1.0f - 2.0f * __builtin_amdgcn_rcpf(e + 1.0f);
}

// One RNN timestep for this wave's (mh, nq) slice. LAST=true: write f32 h_T
// to global instead of f16 h to LDS.
template <bool LAST>
__device__ __forceinline__ void rnn_step(const _Float16* __restrict__ hcur,
                                         _Float16* __restrict__ hnxt,
                                         const _Float16* __restrict__ ecur,
                                         const float* __restrict__ sbias,
                                         const v16h (&whhf)[2][8],
                                         const v16h (&wihf)[2],
                                         float* __restrict__ hout,
                                         int b0, int mh, int nq, int lane) {
#pragma unroll 1
  for (int m = 0; m < 4; ++m) {
    const int row0 = (mh * 4 + m) * 16;
    // A-fragments for this M-tile: emb (K=32) + h (K=256 as 8 tiles)
    v16h ae = lds_fragA(ecur, row0, 32, 0, lane);
    v16h ah[8];
#pragma unroll
    for (int k = 0; k < 8; ++k) ah[k] = lds_fragA(hcur, row0, HDIM, k * 32, lane);

#pragma unroll
    for (int nn = 0; nn < 2; ++nn) {
      const int n0 = (nq * 2 + nn) * 16;
      const int c  = n0 + (lane & 15);
      const float bv = sbias[c];                 // b_ih[n] + b_hh[n]
      v8f acc;
#pragma unroll
      for (int j = 0; j < 8; ++j) acc[j] = bv;

      // x_t contribution: emb_t (16x32, zero-padded) x w_ih^T
      acc = __builtin_amdgcn_wmma_f32_16x16x32_f16(
          false, ae, false, wihf[nn], (short)0, acc, false, false);
      // recurrent contribution: h_{t-1} (16x256) x w_hh^T
#pragma unroll
      for (int k = 0; k < 8; ++k)
        acc = __builtin_amdgcn_wmma_f32_16x16x32_f16(
            false, ah[k], false, whhf[nn][k], (short)0, acc, false, false);

      // D layout: lane 0-15 -> N=lane, M=j ; lane 16-31 -> N=lane-16, M=8+j
      const int rb = row0 + ((lane & 16) ? 8 : 0);
#pragma unroll
      for (int j = 0; j < 8; ++j) {
        const float hv = fast_tanh(acc[j]);
        if (LAST)
          hout[(size_t)(b0 + rb + j) * HDIM + c] = hv;   // final h_T in f32
        else
          hnxt[(rb + j) * HDIM + c] = (_Float16)hv;
      }
    }
  }
}

extern "C" __global__ __launch_bounds__(NTHREADS)
void rnn_fused(const float* __restrict__ in,      // (B, 120, 63)
               const float* __restrict__ conv_w,  // (120, 3)
               const float* __restrict__ conv_b,  // (120,)
               const float* __restrict__ w_ih,    // (256, 21)
               const float* __restrict__ w_hh,    // (256, 256)
               const float* __restrict__ b_ih,    // (256,)
               const float* __restrict__ b_hh,    // (256,)
               float* __restrict__ hout) {        // (B, 256) pre-softmax h_T
  extern __shared__ char smem[];
  _Float16* hbuf  = reinterpret_cast<_Float16*>(smem);            // [2][BT][HDIM]
  _Float16* ebuf  = reinterpret_cast<_Float16*>(smem + 131072);   // [2][BT][32]
  float*    sbias = reinterpret_cast<float*>(smem + 147456);      // [HDIM]

  const int tid  = threadIdx.x;
  const int lane = tid & 31;
  const int wid  = tid >> 5;       // 0..15
  const int mh   = wid >> 3;       // 0..1 : owns M-tiles mh*4 .. mh*4+3
  const int nq   = wid & 7;        // 0..7 : owns N-tiles nq*2, nq*2+1
  const int b0   = blockIdx.x * BT;

  // ---- Resident weight fragments (held in VGPRs for all 120 steps) ----
  v16h whhf[2][8];
  v16h wihf[2];
#pragma unroll
  for (int nn = 0; nn < 2; ++nn) {
    const int n0 = (nq * 2 + nn) * 16;
    wihf[nn] = glob_fragB_wih(w_ih, n0, lane);
#pragma unroll
    for (int k = 0; k < 8; ++k)
      whhf[nn][k] = glob_fragB_whh(w_hh, n0, k * 32, lane);
  }

  // ---- Prologue: zero h0 and emb buffers, stage bias ----
  for (int i = tid; i < BT * HDIM; i += NTHREADS) hbuf[i] = (_Float16)0.0f;
  for (int i = tid; i < 2 * BT * 32; i += NTHREADS) ebuf[i] = (_Float16)0.0f;
  for (int i = tid; i < HDIM; i += NTHREADS) sbias[i] = b_ih[i] + b_hh[i];
  __syncthreads();

  // emb(0): relu(conv) for slice t=0, written into ebuf[0]
  {
    const float cw0 = conv_w[0], cw1 = conv_w[1], cw2 = conv_w[2];
    const float cb  = conv_b[0];
    for (int idx = tid; idx < BT * KIN; idx += NTHREADS) {
      const int r = idx / KIN, i = idx % KIN;
      const float* xp = in + ((size_t)(b0 + r) * T_SEQ + 0) * 63;
      float v = fmaf(cw0, xp[i], fmaf(cw1, xp[21 + i], fmaf(cw2, xp[42 + i], cb)));
      ebuf[r * 32 + i] = (_Float16)fmaxf(v, 0.0f);
    }
  }
  __syncthreads();

  // ---- Steady-state time loop (no global stores, one barrier per step) ----
#pragma unroll 1
  for (int t = 0; t < T_SEQ - 1; ++t) {
    const _Float16* hcur = hbuf + (t & 1) * (BT * HDIM);
    _Float16*       hnxt = hbuf + ((t + 1) & 1) * (BT * HDIM);
    const _Float16* ecur = ebuf + (t & 1) * (BT * 32);
    _Float16*       enxt = ebuf + ((t + 1) & 1) * (BT * 32);

    rnn_step<false>(hcur, hnxt, ecur, sbias, whhf, wihf, hout, b0, mh, nq, lane);

    // Prefetch emb(t+1) into the other buffer while this step's reads are done
    {
      const float cw0 = conv_w[(t + 1) * 3 + 0];
      const float cw1 = conv_w[(t + 1) * 3 + 1];
      const float cw2 = conv_w[(t + 1) * 3 + 2];
      const float cb  = conv_b[t + 1];
      for (int idx = tid; idx < BT * KIN; idx += NTHREADS) {
        const int r = idx / KIN, i = idx % KIN;
        const float* xp = in + ((size_t)(b0 + r) * T_SEQ + (t + 1)) * 63;
        float v = fmaf(cw0, xp[i], fmaf(cw1, xp[21 + i], fmaf(cw2, xp[42 + i], cb)));
        enxt[r * 32 + i] = (_Float16)fmaxf(v, 0.0f);
      }
    }
    __syncthreads();
  }

  // ---- Peeled final step: tanh result goes straight to global in f32 ----
  {
    const int t = T_SEQ - 1;
    const _Float16* hcur = hbuf + (t & 1) * (BT * HDIM);
    const _Float16* ecur = ebuf + (t & 1) * (BT * 32);
    rnn_step<true>(hcur, nullptr, ecur, sbias, whhf, wihf, hout, b0, mh, nq, lane);
  }
}

// Row softmax over HDIM=256 features; one thread per batch row.
extern "C" __global__ void softmax_rows(const float* __restrict__ hin,
                                        float* __restrict__ out, int rows) {
  const int r = blockIdx.x * blockDim.x + threadIdx.x;
  if (r >= rows) return;
  const float* p = hin + (size_t)r * HDIM;
  float m = -INFINITY;
  for (int n = 0; n < HDIM; ++n) m = fmaxf(m, p[n]);
  float s = 0.0f;
  for (int n = 0; n < HDIM; ++n) s += __expf(p[n] - m);
  const float inv = __builtin_amdgcn_rcpf(s);
  float* o = out + (size_t)r * HDIM;
  for (int n = 0; n < HDIM; ++n) o[n] = __expf(p[n] - m) * inv;
}

extern "C" void kernel_launch(void* const* d_in, const int* in_sizes, int n_in,
                              void* d_out, int out_size, void* d_ws, size_t ws_size,
                              hipStream_t stream) {
  (void)n_in; (void)out_size; (void)ws_size;
  const float* in     = (const float*)d_in[0];
  const float* conv_w = (const float*)d_in[1];
  const float* conv_b = (const float*)d_in[2];
  const float* w_ih   = (const float*)d_in[3];
  const float* w_hh   = (const float*)d_in[4];
  const float* b_ih   = (const float*)d_in[5];
  const float* b_hh   = (const float*)d_in[6];
  float* ws  = (float*)d_ws;    // 8192*256 f32 pre-softmax h_T (8 MB)
  float* out = (float*)d_out;

  const int B    = in_sizes[0] / (T_SEQ * 63);   // 8192
  const int nblk = B / BT;                       // 64 workgroups

  // >64KB dynamic LDS: raise the limit (host-side, deterministic, capture-safe)
  (void)hipFuncSetAttribute((const void*)rnn_fused,
                            hipFuncAttributeMaxDynamicSharedMemorySize,
                            SMEM_BYTES);

  rnn_fused<<<nblk, NTHREADS, SMEM_BYTES, stream>>>(
      in, conv_w, conv_b, w_ih, w_hh, b_ih, b_hh, ws);

  softmax_rows<<<(B + 255) / 256, 256, 0, stream>>>(ws, out, B);
}
